// mySelfCorrelationComputation_52621939310858
// MI455X (gfx1250) — matrix operand order repaired and checked by reading, not compiled, not run on hardware
//
#include <hip/hip_runtime.h>
#include <hip/hip_bf16.h>

typedef __attribute__((ext_vector_type(16))) __bf16 v16bf;
typedef __attribute__((ext_vector_type(8)))  __bf16 v8bf;
typedef __attribute__((ext_vector_type(8)))  float  v8f;
typedef __attribute__((ext_vector_type(4)))  int    v4i;

typedef __attribute__((address_space(1))) v4i* gptr_v4i;   // global (prints as __device__)
typedef __attribute__((address_space(3))) v4i* lptr_v4i;   // LDS

// ---------------- bf16 helpers (bit-exact RNE convert) ----------------
__device__ __forceinline__ __bf16 f2bf(float f) {
  unsigned u = __builtin_bit_cast(unsigned, f);
  unsigned r = (u + 0x7FFFu + ((u >> 16) & 1u)) >> 16;
  unsigned short s = (unsigned short)r;
  return __builtin_bit_cast(__bf16, s);
}
__device__ __forceinline__ float bf2f(__bf16 b) {
  unsigned short s = __builtin_bit_cast(unsigned short, b);
  unsigned u = ((unsigned)s) << 16;
  return __builtin_bit_cast(float, u);
}

// ---------------- WMMA fragment loads (wave32, 16x16x32 bf16) ----------------
// A (16xK tile, row-major, ld in elements): lane l holds row l&15,
// K {kb..kb+7} and {kb+16..kb+23}, kb = (l>>4)*8   (ISA 7.12.2)
__device__ __forceinline__ v16bf load_a_frag(const __bf16* tile, int ld) {
  int lane = threadIdx.x & 31;
  int row  = lane & 15;
  int kb   = (lane >> 4) << 3;
  v8bf a = *(const v8bf*)(tile + (size_t)row * ld + kb);
  v8bf b = *(const v8bf*)(tile + (size_t)row * ld + kb + 16);
  return __builtin_shufflevector(a, b, 0,1,2,3,4,5,6,7,8,9,10,11,12,13,14,15);
}
// B (Kx16 tile) supplied transposed (row-major W[o][c]); lane l = column l&15,
// K {kb..kb+7} and {kb+16..kb+23}, kb = (l>>4)*8 (same as A).
__device__ __forceinline__ v16bf load_b_frag(const __bf16* wt, int ld) {
  int lane = threadIdx.x & 31;
  int col  = lane & 15;
  int kb   = (lane >> 4) << 3;
  v8bf a = *(const v8bf*)(wt + (size_t)col * ld + kb);
  v8bf b = *(const v8bf*)(wt + (size_t)col * ld + kb + 16);
  return __builtin_shufflevector(a, b, 0,1,2,3,4,5,6,7,8,9,10,11,12,13,14,15);
}
#define WMMA_BF16(A, B, C) \
  __builtin_amdgcn_wmma_f32_16x16x32_bf16(false, (A), false, (B), (short)0, (C), false, false)

// ---------------- problem constants ----------------
static constexpr int B_    = 16;
static constexpr int HW    = 1024;      // 32*32
static constexpr int NPIX  = B_ * HW;   // 16384
static constexpr int C_IN  = 640;
static constexpr int C_MID = 64;
static constexpr int C_FT  = 1664;
static constexpr int C_OUT = 640;
static constexpr int BK    = 64;        // K-chunk staged in LDS for big GEMMs

// ---------------- workspace layout (bytes; e overlaps xact) ----------------
static constexpr size_t OFF_XACT = 0;                                    // 16384*640 bf16
static constexpr size_t OFF_E    = OFF_XACT;                             // reuse (dead after gemm1)
static constexpr size_t OFF_ID   = OFF_XACT + (size_t)NPIX * C_IN * 2;
static constexpr size_t OFF_FEAT = OFF_ID   + (size_t)NPIX * C_MID * 2;
static constexpr size_t OFF_WIN  = OFF_FEAT + (size_t)NPIX * C_FT * 2;
static constexpr size_t OFF_WEMB = OFF_WIN  + (size_t)C_MID * C_IN * 2;
static constexpr size_t OFF_WOUT = OFF_WEMB + (size_t)C_OUT * C_FT * 2;
static constexpr size_t OFF_SC1  = OFF_WOUT + (size_t)C_OUT * C_OUT * 2;
static constexpr size_t OFF_BI1  = OFF_SC1  + 640 * 4;
static constexpr size_t OFF_SCIN = OFF_BI1  + 640 * 4;
static constexpr size_t OFF_BIIN = OFF_SCIN + 64 * 4;
static constexpr size_t OFF_SCE  = OFF_BIIN + 64 * 4;
static constexpr size_t OFF_BIE  = OFF_SCE  + 640 * 4;
static constexpr size_t OFF_SCO  = OFF_BIE  + 640 * 4;
static constexpr size_t OFF_BIO  = OFF_SCO  + 640 * 4;

// ---------------- prep: fold BN into scale/bias ----------------
__global__ void bnprep_kernel(const float* __restrict__ g, const float* __restrict__ b,
                              const float* __restrict__ m, const float* __restrict__ v,
                              float* __restrict__ sc, float* __restrict__ bi, int n) {
  int i = blockIdx.x * blockDim.x + threadIdx.x;
  if (i < n) {
    float s = g[i] * __frsqrt_rn(v[i] + 1e-5f);
    sc[i] = s;
    bi[i] = b[i] - m[i] * s;
  }
}

__global__ void cvt_bf16_kernel(const float* __restrict__ w, __bf16* __restrict__ o, int n) {
  int i = blockIdx.x * blockDim.x + threadIdx.x;
  if (i < n) o[i] = f2bf(w[i]);
}

// ---------------- BN1+ReLU, NCHW -> (pixel, channel) bf16 ----------------
__global__ __launch_bounds__(256) void prep_x_kernel(const float* __restrict__ x,
                                                     const float* __restrict__ sc,
                                                     const float* __restrict__ bi,
                                                     __bf16* __restrict__ xact) {
  size_t idx = (size_t)blockIdx.x * 256 + threadIdx.x;   // exact: 16*640*1024 / 256
  int b   = (int)(idx / (C_IN * HW));
  int rem = (int)(idx % (C_IN * HW));
  int c   = rem >> 10;
  int hw  = rem & 1023;
  float v = x[idx] * sc[c] + bi[c];
  v = v > 0.f ? v : 0.f;
  xact[(size_t)(b * HW + hw) * C_IN + c] = f2bf(v);
}

// ---------------- GEMM1: (16384x640)x(640x64) + BN + ReLU + L2 norm -> identity ----------------
__global__ __launch_bounds__(256) void gemm1_identity_kernel(
    const __bf16* __restrict__ xact, const __bf16* __restrict__ w_in,
    const float* __restrict__ sc, const float* __restrict__ bi,
    __bf16* __restrict__ identity) {
  __shared__ float sh[128 * 64];                 // 32 KB of 320 KB LDS
  int wave = threadIdx.x >> 5;
  int lane = threadIdx.x & 31;
  int p0 = blockIdx.x * 128 + wave * 16;

  v8f acc[4] = {};
  for (int k = 0; k < C_IN; k += 32) {
    v16bf a = load_a_frag(xact + (size_t)p0 * C_IN + k, C_IN);
#pragma unroll
    for (int j = 0; j < 4; ++j) {
      v16bf bfrag = load_b_frag(w_in + (size_t)(j * 16) * C_IN + k, C_IN);
      acc[j] = WMMA_BF16(a, bfrag, acc[j]);
    }
  }
  int mofs = (lane >> 4) * 8;
  int ncol = lane & 15;
#pragma unroll
  for (int j = 0; j < 4; ++j) {
    int c = j * 16 + ncol;
    float s = sc[c], bb = bi[c];
#pragma unroll
    for (int i = 0; i < 8; ++i) {
      float v = acc[j][i] * s + bb;
      v = v > 0.f ? v : 0.f;                     // BN_in + ReLU (2nd ReLU is no-op)
      sh[(wave * 16 + i + mofs) * 64 + c] = v;
    }
  }
  __syncthreads();
  // per-pixel L2 normalize over 64 channels: 2 threads per pixel
  int pl = threadIdx.x >> 1;
  int ch0 = (threadIdx.x & 1) * 32;
  float ss = 0.f;
#pragma unroll
  for (int c = 0; c < 64; ++c) { float v = sh[pl * 64 + c]; ss += v * v; }
  float rinv = 1.f / fmaxf(sqrtf(ss), 1e-12f);
  size_t base = (size_t)(blockIdx.x * 128 + pl) * C_MID + ch0;
#pragma unroll
  for (int c = 0; c < 32; ++c)
    identity[base + c] = f2bf(sh[pl * 64 + ch0 + c] * rinv);
}

// ---------------- self-correlation + gs norm + concat -> feat(16384,1664) ----------------
__global__ __launch_bounds__(256) void corr_feat_kernel(const __bf16* __restrict__ id,
                                                        __bf16* __restrict__ feat) {
  int wave = threadIdx.x >> 5;
  int lane = threadIdx.x & 31;
  int p  = blockIdx.x * 8 + wave;                // one wave per pixel
  int hw = p & 1023, h = hw >> 5, wc = hw & 31;
  const __bf16* idp = id + (size_t)p * C_MID;
  float v0 = bf2f(idp[lane]);
  float v1 = bf2f(idp[lane + 32]);
  float pr0[25], pr1[25];
  float s = 0.f;
#pragma unroll
  for (int k = 0; k < 25; ++k) {
    int dh = k / 5 - 2, dw = k % 5 - 2;
    int hh = h + dh, ww = wc + dw;
    float q0 = 0.f, q1 = 0.f;
    if (hh >= 0 && hh < 32 && ww >= 0 && ww < 32) {
      const __bf16* idq = id + (size_t)(p + dh * 32 + dw) * C_MID;
      q0 = bf2f(idq[lane]);
      q1 = bf2f(idq[lane + 32]);
    }
    float a = v0 * q0, b = v1 * q1;
    pr0[k] = a; pr1[k] = b;
    s += a * a + b * b;
  }
#pragma unroll
  for (int off = 16; off > 0; off >>= 1) s += __shfl_xor(s, off, 32);
  float rinv = rsqrtf(s + 1e-6f);
  __bf16* fp = feat + (size_t)p * C_FT;
  fp[lane]      = idp[lane];                     // identity passthrough (64 ch)
  fp[lane + 32] = idp[lane + 32];
#pragma unroll
  for (int k = 0; k < 25; ++k) {                 // gs: channel 64 + c*25 + k (torch Unfold order)
    fp[64 + lane * 25 + k]        = f2bf(pr0[k] * rinv);
    fp[64 + (lane + 32) * 25 + k] = f2bf(pr1[k] * rinv);
  }
}

// ---------------- big-GEMM mainloop: 256(M)x64(N) block, LDS-staged B ----------------
// Per wave: 2 M-tiles x 4 N-tiles (8 f32 accumulators). B chunk (64 rows x BK)
// staged cooperatively & coalesced; async-to-LDS path when toolchain exposes it.
template <int K>
__device__ __forceinline__ void gemm_mainloop(const __bf16* __restrict__ A,
                                              const __bf16* __restrict__ W,
                                              int p0, int n0, __bf16* Bsh,
                                              v8f* acc0, v8f* acc1) {
  for (int k0 = 0; k0 < K; k0 += BK) {
    __syncthreads();                             // previous chunk fully consumed
    const __bf16* src0 = W + (size_t)n0 * K + k0;
#pragma unroll
    for (int u = threadIdx.x; u < 512; u += 256) {       // 64 rows x 128B, 16B units
      int r = u >> 3, sgm = (u & 7) * 8;
      const __bf16* src = src0 + (size_t)r * K + sgm;
      __bf16* dst = Bsh + r * BK + sgm;
#if __has_builtin(__builtin_amdgcn_global_load_async_to_lds_b128)
      __builtin_amdgcn_global_load_async_to_lds_b128(
          (gptr_v4i)(unsigned long long)src,
          (lptr_v4i)(unsigned)(unsigned long long)dst, 0, 0);
#else
      *(v8bf*)dst = *(const v8bf*)src;
#endif
    }
#if __has_builtin(__builtin_amdgcn_global_load_async_to_lds_b128)
#if __has_builtin(__builtin_amdgcn_s_wait_asynccnt)
    __builtin_amdgcn_s_wait_asynccnt(0);
#else
    asm volatile("s_wait_asynccnt 0x0" ::: "memory");
#endif
#endif
    __syncthreads();
#pragma unroll
    for (int kk = 0; kk < BK; kk += 32) {
      v16bf a0 = load_a_frag(A + (size_t)p0 * K + k0 + kk, K);
      v16bf a1 = load_a_frag(A + (size_t)(p0 + 16) * K + k0 + kk, K);
#pragma unroll
      for (int j = 0; j < 4; ++j) {
        v16bf bfrag = load_b_frag(Bsh + (j * 16) * BK + kk, BK);   // ds_load_b128 x2
        acc0[j] = WMMA_BF16(a0, bfrag, acc0[j]);
        acc1[j] = WMMA_BF16(a1, bfrag, acc1[j]);
      }
    }
  }
}

// ---------------- GEMM2: (16384x1664)x(1664x640) + BN + ReLU -> e (bf16, pixel-major) ----------------
__global__ __launch_bounds__(256) void gemm_emb_kernel(
    const __bf16* __restrict__ feat, const __bf16* __restrict__ w,
    const float* __restrict__ sc, const float* __restrict__ bi,
    __bf16* __restrict__ e) {
  __shared__ alignas(16) __bf16 Bsh[64 * BK];    // 8 KB
  int wave = threadIdx.x >> 5, lane = threadIdx.x & 31;
  int p0 = blockIdx.x * 256 + wave * 32;
  int n0 = blockIdx.y * 64;
  v8f acc0[4] = {}, acc1[4] = {};
  gemm_mainloop<C_FT>(feat, w, p0, n0, Bsh, acc0, acc1);

  int mofs = (lane >> 4) * 8;
  int ncol = lane & 15;
#pragma unroll
  for (int j = 0; j < 4; ++j) {
    int oc = n0 + j * 16 + ncol;
    float s = sc[oc], bb = bi[oc];
#pragma unroll
    for (int i = 0; i < 8; ++i) {
      float v0 = acc0[j][i] * s + bb;
      float v1 = acc1[j][i] * s + bb;
      v0 = v0 > 0.f ? v0 : 0.f;
      v1 = v1 > 0.f ? v1 : 0.f;
      e[(size_t)(p0 + i + mofs) * C_OUT + oc]      = f2bf(v0);
      e[(size_t)(p0 + 16 + i + mofs) * C_OUT + oc] = f2bf(v1);
    }
  }
}

// ---------------- GEMM3: (16384x640)x(640x640) + BN -> out (fp32, NCHW) ----------------
__global__ __launch_bounds__(256) void gemm_out_kernel(
    const __bf16* __restrict__ e, const __bf16* __restrict__ w,
    const float* __restrict__ sc, const float* __restrict__ bi,
    float* __restrict__ out) {
  __shared__ alignas(16) __bf16 Bsh[64 * BK];
  int wave = threadIdx.x >> 5, lane = threadIdx.x & 31;
  int p0 = blockIdx.x * 256 + wave * 32;
  int n0 = blockIdx.y * 64;
  v8f acc0[4] = {}, acc1[4] = {};
  gemm_mainloop<C_OUT>(e, w, p0, n0, Bsh, acc0, acc1);

  int mofs = (lane >> 4) * 8;
  int ncol = lane & 15;
#pragma unroll
  for (int j = 0; j < 4; ++j) {
    int oc = n0 + j * 16 + ncol;
    float s = sc[oc], bb = bi[oc];
#pragma unroll
    for (int i = 0; i < 8; ++i) {
      int pa = p0 + i + mofs;
      int pb = pa + 16;
      out[((size_t)(pa >> 10) * C_OUT + oc) * HW + (pa & 1023)] = acc0[j][i] * s + bb;
      out[((size_t)(pb >> 10) * C_OUT + oc) * HW + (pb & 1023)] = acc1[j][i] * s + bb;
    }
  }
}

// ---------------- launch ----------------
extern "C" void kernel_launch(void* const* d_in, const int* in_sizes, int n_in,
                              void* d_out, int out_size, void* d_ws, size_t ws_size,
                              hipStream_t stream) {
  const float* x      = (const float*)d_in[0];
  const float* bn1_g  = (const float*)d_in[1];
  const float* bn1_b  = (const float*)d_in[2];
  const float* bn1_m  = (const float*)d_in[3];
  const float* bn1_v  = (const float*)d_in[4];
  const float* w_in   = (const float*)d_in[5];
  const float* bnin_g = (const float*)d_in[6];
  const float* bnin_b = (const float*)d_in[7];
  const float* bnin_m = (const float*)d_in[8];
  const float* bnin_v = (const float*)d_in[9];
  const float* w_emb  = (const float*)d_in[10];
  const float* bne_g  = (const float*)d_in[11];
  const float* bne_b  = (const float*)d_in[12];
  const float* bne_m  = (const float*)d_in[13];
  const float* bne_v  = (const float*)d_in[14];
  const float* w_out  = (const float*)d_in[15];
  const float* bno_g  = (const float*)d_in[16];
  const float* bno_b  = (const float*)d_in[17];
  const float* bno_m  = (const float*)d_in[18];
  const float* bno_v  = (const float*)d_in[19];

  char* ws = (char*)d_ws;
  __bf16* xact  = (__bf16*)(ws + OFF_XACT);
  __bf16* ident = (__bf16*)(ws + OFF_ID);
  __bf16* feat  = (__bf16*)(ws + OFF_FEAT);
  __bf16* ebuf  = (__bf16*)(ws + OFF_E);
  __bf16* winb  = (__bf16*)(ws + OFF_WIN);
  __bf16* wembb = (__bf16*)(ws + OFF_WEMB);
  __bf16* woutb = (__bf16*)(ws + OFF_WOUT);
  float* sc1  = (float*)(ws + OFF_SC1),  *bi1  = (float*)(ws + OFF_BI1);
  float* scin = (float*)(ws + OFF_SCIN), *biin = (float*)(ws + OFF_BIIN);
  float* sce  = (float*)(ws + OFF_SCE),  *bie  = (float*)(ws + OFF_BIE);
  float* sco  = (float*)(ws + OFF_SCO),  *bio  = (float*)(ws + OFF_BIO);

  bnprep_kernel<<<3, 256, 0, stream>>>(bn1_g, bn1_b, bn1_m, bn1_v, sc1, bi1, 640);
  bnprep_kernel<<<1, 256, 0, stream>>>(bnin_g, bnin_b, bnin_m, bnin_v, scin, biin, 64);
  bnprep_kernel<<<3, 256, 0, stream>>>(bne_g, bne_b, bne_m, bne_v, sce, bie, 640);
  bnprep_kernel<<<3, 256, 0, stream>>>(bno_g, bno_b, bno_m, bno_v, sco, bio, 640);

  cvt_bf16_kernel<<<(C_MID * C_IN + 255) / 256, 256, 0, stream>>>(w_in, winb, C_MID * C_IN);
  cvt_bf16_kernel<<<(C_OUT * C_FT + 255) / 256, 256, 0, stream>>>(w_emb, wembb, C_OUT * C_FT);
  cvt_bf16_kernel<<<(C_OUT * C_OUT + 255) / 256, 256, 0, stream>>>(w_out, woutb, C_OUT * C_OUT);

  prep_x_kernel<<<(NPIX * C_IN) / 256, 256, 0, stream>>>(x, sc1, bi1, xact);
  gemm1_identity_kernel<<<NPIX / 128, 256, 0, stream>>>(xact, winb, scin, biin, ident);
  corr_feat_kernel<<<NPIX / 8, 256, 0, stream>>>(ident, feat);
  gemm_emb_kernel<<<dim3(NPIX / 256, C_OUT / 64), 256, 0, stream>>>(feat, wembb, sce, bie, ebuf);
  gemm_out_kernel<<<dim3(NPIX / 256, C_OUT / 64), 256, 0, stream>>>(ebuf, woutb, sco, bio, (float*)d_out);
}
